// ConceptLayer_86036784873949
// MI455X (gfx1250) — matrix operand order repaired and checked by reading, not compiled, not run on hardware
//
#include <hip/hip_runtime.h>
#include <hip/hip_bf16.h>

// ---------------------------------------------------------------------------
// ConceptLayer on MI455X (gfx1250, wave32, WMMA bf16 + TDM tile staging)
// B=8, T=2048, E=512, H=16, D=32.  M = B*T = 16384 token rows.
// ---------------------------------------------------------------------------

typedef __attribute__((ext_vector_type(16))) __bf16 v16bf;
typedef __attribute__((ext_vector_type(8)))  float  v8f;
typedef __attribute__((ext_vector_type(4)))  unsigned int uint4v;
typedef __attribute__((ext_vector_type(8)))  int    int8v;
typedef __attribute__((ext_vector_type(4)))  int    int4v;
typedef unsigned short u16;
typedef unsigned int   u32;

#define Bn 8
#define Tn 2048
#define En 512
#define Hn 16
#define Dn 32
#define Mrows (Bn * Tn)            // 16384
#define SCAN_CHUNKS 16
#define SCAN_L 128                 // Tn / SCAN_CHUNKS

#if __has_builtin(__builtin_amdgcn_tensor_load_to_lds)
#define HAVE_TDM 1
#else
#define HAVE_TDM 0
#endif

__device__ __constant__ float INV_DECAY   = 0.83333333333333337f;   // 1/1.2
__device__ __constant__ float ALPHA_L     = 7.3248e-11f;            // (1/1.2)^128
__device__ __constant__ float LOG2_ALPHA  = -0.26303440583379383f;  // log2(1/1.2)

// native fp32 -> bf16 (RNE)
__device__ inline u16 f2bf(float f) {
  __bf16 h = (__bf16)f;
  return __builtin_bit_cast(u16, h);
}
// bf16 -> fp32 is exact: shift into the high half
__device__ inline float bf2f(u16 u) {
  return __uint_as_float(((u32)u) << 16);
}
// packed bf16 multiply (VOP3P op 42)
__device__ inline u32 pk_mul_bf16(u32 a, u32 b) {
  u32 d;
  asm("v_pk_mul_bf16 %0, %1, %2" : "=v"(d) : "v"(a), "v"(b));
  return d;
}

union FragBF {
  v16bf v;
  u32 u[8];
};

__device__ inline v8f wmma_bf16(const FragBF& a, const FragBF& b, v8f c) {
  // D = A(16x32 bf16) * B(32x16 bf16) + C(16x16 f32)
  return __builtin_amdgcn_wmma_f32_16x16x32_bf16(
      false, a.v, false, b.v, (short)0, c, false, false);
}

#if HAVE_TDM
// ---------------------------------------------------------------------------
// TDM: DMA a 2D tile (16-bit elements) from global into LDS, row-major packed.
// D# per CDNA5 ISA: group0 = {count/type/addr},
// group1 = {data_size, tensor dims, tile dims, dim0 stride}; 2D -> g2=g3=0.
// Issued once per workgroup (TDM ignores EXEC); tracked with TENSORcnt.
// This toolchain's builtin takes 6 args (g0, g1, g2, g3, extra int32x8, cpol).
// ---------------------------------------------------------------------------
__device__ inline void tdm_load_2d_b16(u32 lds_byte_addr, const void* gptr,
                                       u32 tensor_d0, u32 tensor_d1,
                                       u32 tile_d0, u32 tile_d1, u32 stride0) {
  unsigned long long ga = (unsigned long long)gptr;
  uint4v g0;
  g0[0] = 1u;                                              // count=1 (user D#)
  g0[1] = lds_byte_addr;                                   // lds_addr [63:32]
  g0[2] = (u32)ga;                                         // global_addr lo
  g0[3] = (u32)((ga >> 32) & 0x1FFFFFFu) | 0x80000000u;    // addr hi | type=2
  int8v g1;
  g1[0] = (int)(1u << 16);                                 // data_size = 2B
  g1[1] = (int)(tensor_d0 << 16);                          // tensor_dim0 lo
  g1[2] = (int)((tensor_d0 >> 16) | (tensor_d1 << 16));    // td0 hi | td1 lo
  g1[3] = (int)((tensor_d1 >> 16) | (tile_d0 << 16));      // td1 hi | tile0
  g1[4] = (int)(tile_d1 & 0xFFFFu);                        // tile1 | tile2=0
  g1[5] = (int)stride0;                                    // dim0 stride lo
  g1[6] = 0;                                               // stride hi | s1 lo
  g1[7] = 0;
  int4v z4 = {0, 0, 0, 0};
  int8v z8 = {0, 0, 0, 0, 0, 0, 0, 0};
  __builtin_amdgcn_tensor_load_to_lds(g0, g1, z4, z4, z8, 0);
}
__device__ inline u32 lds_addr_of(const void* p) {
  // generic shared pointer: low 32 bits are the LDS byte address
  return (u32)(size_t)p;
}
#endif

// ---------------------------------------------------------------------------
// Kernel 0a: vectorized fp32 -> packed bf16 (x and concept_map)
// ---------------------------------------------------------------------------
__global__ __launch_bounds__(256)
void cvt4(const float4* __restrict__ in, uint2* __restrict__ out, int n4) {
  int i = blockIdx.x * 256 + threadIdx.x;
  if (i < n4) {
    float4 v = in[i];
    uint2 o;
    o.x = (u32)f2bf(v.x) | ((u32)f2bf(v.y) << 16);
    o.y = (u32)f2bf(v.z) | ((u32)f2bf(v.w) << 16);
    out[i] = o;
  }
}

// ---------------------------------------------------------------------------
// Kernel 0b: W (k-major) -> Wt bf16 (n-major)
// ---------------------------------------------------------------------------
__global__ __launch_bounds__(256)
void cvt_w_transpose(const float* __restrict__ W, u16* __restrict__ Wt) {
  int i = blockIdx.x * 256 + threadIdx.x;   // 262144
  int n = i >> 9, k = i & 511;
  Wt[i] = f2bf(W[(size_t)k * En + n]);
}

// ---------------------------------------------------------------------------
// Kernel 1: h = x @ W + b.  128x64 block tile, 256 threads (8 waves).
// Tiles staged into LDS by the Tensor Data Mover (double buffered); one
// s_wait_tensorcnt + barrier per K-step; WMMA bf16 with fp32 accumulate.
// ---------------------------------------------------------------------------
__global__ __launch_bounds__(256)
void proj_gemm(const u16* __restrict__ xbf, const u16* __restrict__ Wt,
               const float* __restrict__ bias, float* __restrict__ hout) {
  __shared__ u16 sA[2][128 * 32];  // [row][k]
  __shared__ u16 sB[2][64 * 32];   // [n][k]

  const int m0 = blockIdx.x * 128;
  const int n0 = blockIdx.y * 64;
  const int tid = threadIdx.x;
  const int wave = tid >> 5;
  const int lane = tid & 31;

#if HAVE_TDM
  auto stage = [&](int it, int buf) {
    if (tid == 0) {
      // A tile: rows m0..m0+127, k-cols it*32..+31 of 16384x512 bf16 tensor
      tdm_load_2d_b16(lds_addr_of(&sA[buf][0]),
                      xbf + (size_t)m0 * En + it * 32,
                      En, Mrows, 32, 128, En);
      // B tile: rows n0..n0+63, k-cols it*32..+31 of 512x512 bf16 tensor
      tdm_load_2d_b16(lds_addr_of(&sB[buf][0]),
                      Wt + (size_t)n0 * En + it * 32,
                      En, En, 32, 64, En);
    }
  };
  auto wait_tiles = [&]() {
    if (tid == 0) __builtin_amdgcn_s_wait_tensorcnt((short)0);
  };
#else
  auto stage = [&](int it, int buf) {
    const int kd = it * 16;
    u32* dA = (u32*)sA[buf];
    const u32* gx = (const u32*)xbf;
#pragma unroll
    for (int j = 0; j < 8; ++j) {
      int d = tid + j * 256;
      int r = d >> 4, c = d & 15;
      dA[d] = gx[(size_t)(m0 + r) * 256 + kd + c];
    }
    u32* dB = (u32*)sB[buf];
    const u32* gw = (const u32*)Wt;
#pragma unroll
    for (int j = 0; j < 4; ++j) {
      int d = tid + j * 256;
      int r = d >> 4, c = d & 15;
      dB[d] = gw[(size_t)(n0 + r) * 256 + kd + c];
    }
  };
  auto wait_tiles = [&]() {};
#endif

  v8f acc[4] = {};
  stage(0, 0);

  for (int it = 0; it < 16; ++it) {
    wait_tiles();          // issuing wave drains TENSORcnt before signaling
    __syncthreads();
    if (it + 1 < 16) stage(it + 1, (it + 1) & 1);

    const u16* A  = sA[it & 1];
    const u16* Bt = sB[it & 1];

    // A fragment: lane = M row; K striped per ISA 16-bit A layout
    FragBF a;
    const int mrow = wave * 16 + (lane & 15);
    const int kbase = (lane >> 4) * 8;
#pragma unroll
    for (int v = 0; v < 8; ++v) {
      int k = kbase + (v & 3) * 2 + (v >> 2) * 16;
      a.u[v] = *(const u32*)&A[mrow * 32 + k];
    }
#pragma unroll
    for (int nt = 0; nt < 4; ++nt) {
      FragBF bfrag;
      const int n = nt * 16 + (lane & 15);
      const int kb2 = (lane >> 4) * 8;
#pragma unroll
      for (int v = 0; v < 8; ++v) {
        bfrag.u[v] = *(const u32*)&Bt[n * 32 + kb2 + (v & 3) * 2 + (v >> 2) * 16];
      }
      acc[nt] = wmma_bf16(a, bfrag, acc[nt]);
    }
  }

  // C/D layout: VGPR r -> M=r (lanes 0-15), M=r+8 (lanes 16-31)
  const int mloc = wave * 16 + ((lane >> 4) ? 8 : 0);
#pragma unroll
  for (int nt = 0; nt < 4; ++nt) {
#pragma unroll
    for (int r = 0; r < 8; ++r) {
      int m = m0 + mloc + r;
      int n = n0 + nt * 16 + (lane & 15);
      hout[(size_t)m * En + n] = acc[nt][r] + bias[n];
    }
  }
}

// ---------------------------------------------------------------------------
// Kernel 2a: chunk-local decayed scan.  emit S_t = s; s' = (s + h_t)/1.2
// ---------------------------------------------------------------------------
__global__ __launch_bounds__(256)
void scan_local(const float* __restrict__ h, float* __restrict__ Sl,
                float* __restrict__ carry) {
  const int tid = blockIdx.x * 256 + threadIdx.x;  // 65536
  const int e = tid & 511;
  const int c = (tid >> 9) & 15;
  const int b = tid >> 13;
  size_t idx = ((size_t)(b * Tn + c * SCAN_L)) * En + e;
  float s = 0.f;
  for (int tl = 0; tl < SCAN_L; ++tl) {
    float hv = h[idx];
    Sl[idx] = s;
    s = (s + hv) * INV_DECAY;
    idx += En;
  }
  carry[((size_t)(b * En + e)) * SCAN_CHUNKS + c] = s;
}

// ---------------------------------------------------------------------------
// Kernel 2b: combine carries: s_in(c+1) = s_in(c)*alpha^L + carry(c)
// ---------------------------------------------------------------------------
__global__ __launch_bounds__(256)
void scan_combine(const float* __restrict__ carry, float* __restrict__ inc) {
  const int ch = blockIdx.x * 256 + threadIdx.x;  // 4096
  float s = 0.f;
  for (int c = 0; c < SCAN_CHUNKS; ++c) {
    inc[(size_t)ch * SCAN_CHUNKS + c] = s;
    s = s * ALPHA_L + carry[(size_t)ch * SCAN_CHUNKS + c];
  }
}

// ---------------------------------------------------------------------------
// Kernel 3: per-head bilinear GEMM (K=1024, N=32) + residual + LayerNorm.
// Block = 32 tokens, 256 threads (8 waves); wave w owns heads {w, w+8}.
// A frag at K-block kb = h[m][kb] (scalar) * 16-elem S slice:
//   1 ds_load_u16 + 2 ds_load_b128 + 8 v_pk_mul_bf16.
// LDS: h,S packed bf16 (32KB each) + y fp32 (64KB) = 128 KB -> 2 blocks/WGP.
// ---------------------------------------------------------------------------
__global__ __launch_bounds__(256)
void bilinear_ln(const float* __restrict__ hbuf, const float* __restrict__ Sl,
                 const float* __restrict__ inc, const u16* __restrict__ Cbf,
                 const float* __restrict__ gamma, const float* __restrict__ beta,
                 float* __restrict__ out) {
  extern __shared__ char smem[];
  u16*   shb = (u16*)smem;                       // 32*512 bf16
  u16*   sSb = (u16*)(smem + 32 * En * 2);       // 32*512 bf16
  float* sy  = (float*)(smem + 2 * 32 * En * 2); // 32*512 fp32

  const int tid = threadIdx.x;
  const int wave = tid >> 5;
  const int lane = tid & 31;
  const long g0 = (long)blockIdx.x * 32;   // same batch b for all 32 tokens
  const int b = (int)(g0 >> 11);
  const int t0 = (int)(g0 & (Tn - 1));

  // load h, S (with scan carry fixup), convert to packed bf16
  for (int i = tid; i < 32 * En; i += 256) {
    int m = i >> 9, e = i & 511;
    long gi = (g0 + m) * En + e;
    shb[i] = f2bf(hbuf[gi]);
    int t = t0 + m;
    float fix = inc[((size_t)(b * En + e)) * SCAN_CHUNKS + (t >> 7)] *
                exp2f((float)(t & (SCAN_L - 1)) * LOG2_ALPHA);
    sSb[i] = f2bf(Sl[gi] + fix);
  }
  __syncthreads();

  for (int head = wave; head < Hn; head += 8) {
    v8f acc[2][2] = {};  // [mtile][ntile]
    for (int kb = 0; kb < 32; ++kb) {      // K = 1024 in steps of 32; ii == kb
      FragBF a[2];
      const int kbase = (lane >> 4) * 8;
#pragma unroll
      for (int mt = 0; mt < 2; ++mt) {
        const int m = mt * 16 + (lane & 15);
        const u16 hs = shb[m * En + head * Dn + kb];
        const u32 hd = (u32)hs * 0x10001u;  // duplicate into both halves
#pragma unroll
        for (int v = 0; v < 8; ++v) {
          int jj = kbase + (v & 3) * 2 + (v >> 2) * 16;   // even
          u32 sp = *(const u32*)&sSb[m * En + head * Dn + jj];
          a[mt].u[v] = pk_mul_bf16(hd, sp);
        }
      }
#pragma unroll
      for (int nt = 0; nt < 2; ++nt) {
        FragBF bfrag;
        const int n = nt * 16 + (lane & 15);
        const int kb2 = (lane >> 4) * 8;
#pragma unroll
        for (int v = 0; v < 8; ++v) {
          int jj = kb2 + (v & 3) * 2 + (v >> 2) * 16;  // two 16-byte runs per lane
          bfrag.u[v] = *(const u32*)
              &Cbf[(((size_t)head * Dn + n) * Dn + kb) * Dn + jj];
        }
        acc[0][nt] = wmma_bf16(a[0], bfrag, acc[0][nt]);
        acc[1][nt] = wmma_bf16(a[1], bfrag, acc[1][nt]);
      }
    }
    // epilogue: y = bilinear + h residual -> LDS
#pragma unroll
    for (int mt = 0; mt < 2; ++mt) {
#pragma unroll
      for (int nt = 0; nt < 2; ++nt) {
#pragma unroll
        for (int r = 0; r < 8; ++r) {
          int m = mt * 16 + r + ((lane >> 4) ? 8 : 0);
          int e = head * Dn + nt * 16 + (lane & 15);
          sy[m * En + e] = acc[mt][nt][r] + bf2f(shb[m * En + e]);
        }
      }
    }
  }
  __syncthreads();

  // LayerNorm over E=512; wave handles 4 tokens; wave32 shuffle reduction
  for (int m = wave * 4; m < wave * 4 + 4; ++m) {
    float s = 0.f, ss = 0.f;
    for (int e = lane; e < En; e += 32) {
      float v = sy[m * En + e];
      s += v; ss += v * v;
    }
#pragma unroll
    for (int off = 16; off > 0; off >>= 1) {
      s  += __shfl_xor(s, off, 32);
      ss += __shfl_xor(ss, off, 32);
    }
    const float mu = s * (1.f / En);
    const float var = ss * (1.f / En) - mu * mu;
    const float rstd = rsqrtf(var + 1e-3f);
    for (int e = lane; e < En; e += 32) {
      out[(size_t)(g0 + m) * En + e] =
          (sy[m * En + e] - mu) * rstd * gamma[e] + beta[e];
    }
  }
}

// ---------------------------------------------------------------------------
// Launch
// ---------------------------------------------------------------------------
extern "C" void kernel_launch(void* const* d_in, const int* in_sizes, int n_in,
                              void* d_out, int out_size, void* d_ws, size_t ws_size,
                              hipStream_t stream) {
  const float* x     = (const float*)d_in[0];  // (8,2048,512)
  const float* W     = (const float*)d_in[1];  // (512,512)
  const float* bias  = (const float*)d_in[2];  // (512,)
  const float* cmap  = (const float*)d_in[3];  // (16,32,32,32)
  const float* gamma = (const float*)d_in[4];  // (512,)
  const float* beta  = (const float*)d_in[5];  // (512,)
  float* out = (float*)d_out;

  // workspace carve
  char* ws = (char*)d_ws;
  float* hbuf  = (float*)(ws);                        // 32 MB
  float* Sl    = (float*)(ws + (size_t)33554432);     // 32 MB
  u16*   xbf   = (u16*)  (ws + (size_t)67108864);     // 16 MB
  u16*   Wt    = (u16*)  (ws + (size_t)83886080);     // 512 KB
  u16*   Cbf   = (u16*)  (ws + (size_t)84410368);     // 1 MB
  float* carry = (float*)(ws + (size_t)85458944);     // 256 KB
  float* incb  = (float*)(ws + (size_t)85721088);     // 256 KB

  // 0) precision conversions (once per launch)
  cvt4<<<dim3((Mrows * En / 4) / 256), dim3(256), 0, stream>>>(
      (const float4*)x, (uint2*)xbf, Mrows * En / 4);
  cvt4<<<dim3((Hn * Dn * Dn * Dn / 4) / 256), dim3(256), 0, stream>>>(
      (const float4*)cmap, (uint2*)Cbf, Hn * Dn * Dn * Dn / 4);
  cvt_w_transpose<<<dim3((En * En) / 256), dim3(256), 0, stream>>>(W, Wt);

  // 1) projection GEMM (TDM-staged tiles, bf16 WMMA, bias fused)
  proj_gemm<<<dim3(Mrows / 128, En / 64), dim3(256), 0, stream>>>(
      xbf, Wt, bias, hbuf);

  // 2) decayed scan (chunked, 3-phase with fixup fused into kernel 3)
  scan_local<<<dim3(65536 / 256), dim3(256), 0, stream>>>(hbuf, Sl, carry);
  scan_combine<<<dim3(4096 / 256), dim3(256), 0, stream>>>(carry, incb);

  // 3) bilinear per-head GEMM + residual + LayerNorm (128 KB dynamic LDS)
  bilinear_ln<<<dim3(Mrows / 32), dim3(256), 131072, stream>>>(
      hbuf, Sl, incb, Cbf, gamma, beta, out);
}
